// sampleLSTM_AR_13125420057080
// MI455X (gfx1250) — compile-verified, matrix operand order checked by reading, not run on hardware
//
#include <hip/hip_runtime.h>

typedef __attribute__((ext_vector_type(16))) _Float16 v16h;
typedef __attribute__((ext_vector_type(8)))  float    v8f;

#define NBLK 36
#define NTHR 128
#define BATCH 96
#define HID 384
#define G4 1536
#define KP0 416
#define KPA 32
#define KP1 768
#define HOR 12

// ---- workspace layout (bytes), all 16B aligned ----
#define OFF_BAR      0
#define OFF_WCAT0    256
#define SZ_WCAT0     (G4*KP0*2)
#define OFF_WCAT1    (OFF_WCAT0 + SZ_WCAT0)
#define SZ_WCAT1     (G4*KP1*2)
#define OFF_WAR0     (OFF_WCAT1 + SZ_WCAT1)
#define SZ_WAR0      (G4*KPA*2)
#define OFF_B0S      (OFF_WAR0 + SZ_WAR0)
#define OFF_B1S      (OFF_B0S + G4*4)
#define OFF_ZBEG     (OFF_B1S + G4*4)
#define OFF_HC0      OFF_ZBEG
#define SZ_HC0       (2*BATCH*KP0*2)
#define OFF_HC1      (OFF_HC0 + SZ_HC0)
#define SZ_HC1       (2*BATCH*KP1*2)
#define OFF_HAR      (OFF_HC1 + SZ_HC1)
#define SZ_HAR       (BATCH*KPA*2)
#define OFF_HL1      (OFF_HAR + SZ_HAR)
#define SZ_HL1       (BATCH*HID*2)
#define OFF_C0       (OFF_HL1 + SZ_HL1)
#define OFF_C1       (OFF_C0 + BATCH*HID*4)
#define OFF_H1F      (OFF_C1 + BATCH*HID*4)
#define OFF_ZEND     (OFF_H1F + BATCH*HID*4)
#define OFF_RAR      OFF_ZEND
#define OFF_RFUT     (OFF_RAR + BATCH*96*4)

// ---------- helpers ----------
__device__ __forceinline__ float sigm_f(float x) {
  return 1.0f / (1.0f + __expf(-x));
}
__device__ __forceinline__ float tanh_f(float x) {
  x = fminf(fmaxf(x, -20.0f), 20.0f);
  float e = __expf(2.0f * x);
  return (e - 1.0f) / (e + 1.0f);
}

// Load a 16-element f16 WMMA fragment slice (two b128 loads) for one lane.
// Layout per CDNA5 ISA: lane group g selects k-base offset of 8 elements,
// q[0] covers k 0..7(+8g), q[1] covers k 16..23(+8g) within a 32-wide k-tile.
__device__ __forceinline__ v16h load_frag16(const _Float16* p) {
  union { v16h v; uint4 q[2]; } u;
  u.q[0] = *(const uint4*)(p);
  u.q[1] = *(const uint4*)(p + 16);
  return u.v;
}

struct Acc4 { v8f a[4]; };

// Stage the block's 16-row A panel (rows mt*16 .. mt*16+15, contiguous in
// memory) into LDS; all 4 waves of the block share the same mt.
__device__ __forceinline__ void stage_A(_Float16* sA, const _Float16* A,
                                        int AKP, int mt) {
  const uint4* src = (const uint4*)(A + (size_t)mt * 16 * AKP);
  uint4* dst = (uint4*)sA;
  const int n4 = (16 * AKP) / 8;
  for (int i = threadIdx.x; i < n4; i += NTHR) dst[i] = src[i];
  __syncthreads();
}

// One wave computes the 4 gate tiles (i,f,g,o) for output block (mt, jt),
// with the A panel already staged in LDS (row-major, 16 x AKP halves):
// gates[m, gi*384 + jt*16 + n] = sum_k A[m,k] * W[gi*384+jt*16+n, k] + bias
__device__ __forceinline__ Acc4 wave_gemm(const _Float16* __restrict__ As, int AKP,
                                          const _Float16* __restrict__ W, int WKP,
                                          int KT, const float* __restrict__ bias,
                                          int jt) {
  const int lane = threadIdx.x & 31;
  const int g = lane >> 4, l16 = lane & 15;
  const _Float16* ap = As + (size_t)l16 * AKP + g * 8;

  Acc4 r;
#pragma unroll
  for (int gi = 0; gi < 4; ++gi) {
    float b = bias[gi * HID + jt * 16 + l16];
    r.a[gi] = (v8f){b, b, b, b, b, b, b, b};
  }
  for (int kt = 0; kt < KT; ++kt) {
    v16h af = load_frag16(ap + kt * 32);
#pragma unroll
    for (int gi = 0; gi < 4; ++gi) {
      const int wrow = gi * HID + jt * 16 + l16;
      v16h bf = load_frag16(W + (size_t)wrow * WKP + kt * 32 + g * 8);
      r.a[gi] = __builtin_amdgcn_wmma_f32_16x16x32_f16(
          false, af, false, bf, (short)0, r.a[gi], false, false);
    }
  }
  return r;
}

// Full LSTM cell epilogue on in-register accumulators.
__device__ __forceinline__ void lstm_tail(const Acc4& r, int mt, int jt,
                                          float* __restrict__ cstate,
                                          _Float16* dA, int strideA, int offA,
                                          _Float16* dB, int strideB, int offB,
                                          float* h32, int write32) {
  const int lane = threadIdx.x & 31;
  const int g = lane >> 4, l16 = lane & 15;
  const int col = jt * 16 + l16;
#pragma unroll
  for (int rr = 0; rr < 8; ++rr) {
    int m = mt * 16 + g * 8 + rr;
    float iv = r.a[0][rr], fv = r.a[1][rr], gv = r.a[2][rr], ov = r.a[3][rr];
    float cp = cstate[m * HID + col];
    float c = sigm_f(fv) * cp + sigm_f(iv) * tanh_f(gv);
    cstate[m * HID + col] = c;
    float h = sigm_f(ov) * tanh_f(c);
    _Float16 hh = (_Float16)h;
    dA[(size_t)m * strideA + offA + col] = hh;
    if (dB) dB[(size_t)m * strideB + offB + col] = hh;
    if (write32) h32[m * HID + col] = h;
  }
}

// Stateless cell (_cell0): c = sig(i)*tanh(g); h = sig(o)*tanh(c)
__device__ __forceinline__ void cell0_tail(const Acc4& r, int mt, int jt,
                                           _Float16* dA, int strideA,
                                           float* h32) {
  const int lane = threadIdx.x & 31;
  const int g = lane >> 4, l16 = lane & 15;
  const int col = jt * 16 + l16;
#pragma unroll
  for (int rr = 0; rr < 8; ++rr) {
    int m = mt * 16 + g * 8 + rr;
    float iv = r.a[0][rr], gv = r.a[2][rr], ov = r.a[3][rr];
    float c = sigm_f(iv) * tanh_f(gv);
    float h = sigm_f(ov) * tanh_f(c);
    if (dA) dA[(size_t)m * strideA + col] = (_Float16)h;
    if (h32) h32[m * HID + col] = h;
  }
}

// Monotonic software grid barrier (36 co-resident blocks).
__device__ __forceinline__ void gbar(unsigned* cnt, unsigned& phase) {
  ++phase;
  __threadfence();
  __syncthreads();
  if (threadIdx.x == 0) {
    __hip_atomic_fetch_add(cnt, 1u, __ATOMIC_RELEASE, __HIP_MEMORY_SCOPE_AGENT);
    const unsigned target = phase * (unsigned)NBLK;
    while (__hip_atomic_load(cnt, __ATOMIC_ACQUIRE, __HIP_MEMORY_SCOPE_AGENT) < target) {
      __builtin_amdgcn_s_sleep(2);
    }
  }
  __syncthreads();
  __threadfence();
}

// ---------- kernel 1: weight prep (f32 -> f16 concatenated, bias sums) ----------
__global__ void lstm_prep_k(const float* __restrict__ Wih0, const float* __restrict__ Whh0,
                            const float* __restrict__ bih0, const float* __restrict__ bhh0,
                            const float* __restrict__ Wih1, const float* __restrict__ Whh1,
                            const float* __restrict__ bih1, const float* __restrict__ bhh1,
                            char* __restrict__ ws) {
  _Float16* wc0 = (_Float16*)(ws + OFF_WCAT0);
  _Float16* wc1 = (_Float16*)(ws + OFF_WCAT1);
  _Float16* war = (_Float16*)(ws + OFF_WAR0);
  float* b0s = (float*)(ws + OFF_B0S);
  float* b1s = (float*)(ws + OFF_B1S);
  unsigned* bars = (unsigned*)(ws + OFF_BAR);

  const int N0 = G4 * KP0, N1 = G4 * KP1, NA = G4 * KPA;
  const int total = N0 + N1 + NA + G4 + G4 + 16;
  for (int i = blockIdx.x * blockDim.x + threadIdx.x; i < total;
       i += gridDim.x * blockDim.x) {
    if (i < N0) {
      int n = i / KP0, k = i - n * KP0;
      float v = (k < 8) ? Wih0[n * 8 + k] : ((k < 392) ? Whh0[n * HID + (k - 8)] : 0.0f);
      wc0[i] = (_Float16)v;
    } else if (i < N0 + N1) {
      int j = i - N0;
      int n = j / KP1, k = j - n * KP1;
      float v = (k < HID) ? Wih1[n * HID + k] : Whh1[n * HID + (k - HID)];
      wc1[j] = (_Float16)v;
    } else if (i < N0 + N1 + NA) {
      int j = i - N0 - N1;
      int n = j / KPA, k = j - n * KPA;
      war[j] = (_Float16)((k < 8) ? Wih0[n * 8 + k] : 0.0f);
    } else if (i < N0 + N1 + NA + G4) {
      int j = i - N0 - N1 - NA;
      b0s[j] = bih0[j] + bhh0[j];
    } else if (i < N0 + N1 + NA + 2 * G4) {
      int j = i - N0 - N1 - NA - G4;
      b1s[j] = bih1[j] + bhh1[j];
    } else {
      bars[i - (N0 + N1 + NA + 2 * G4)] = 0u;
    }
  }
}

// ---------- kernel 2: autoregressive chain (95 stateless steps) ----------
__global__ void __launch_bounds__(NTHR) lstm_ar_k(const float* __restrict__ inputs,
                                                  const float* __restrict__ W_out,
                                                  const float* __restrict__ b_out,
                                                  float* __restrict__ d_out,
                                                  char* __restrict__ ws, int barSlot) {
  __shared__ __align__(16) _Float16 sA[16 * KP1];

  unsigned* bar = (unsigned*)(ws + OFF_BAR) + barSlot;
  const _Float16* war = (const _Float16*)(ws + OFF_WAR0);
  const _Float16* wc1 = (const _Float16*)(ws + OFF_WCAT1);
  const float* b0s = (const float*)(ws + OFF_B0S);
  const float* b1s = (const float*)(ws + OFF_B1S);
  _Float16* hAR = (_Float16*)(ws + OFF_HAR);
  _Float16* hL1 = (_Float16*)(ws + OFF_HL1);
  float* h1f = (float*)(ws + OFF_H1F);
  float* rAR = (float*)(ws + OFF_RAR);
  float* rFUT = (float*)(ws + OFF_RFUT);

  const int tid = threadIdx.x;
  const int wave = blockIdx.x * 4 + (tid >> 5);
  const int mt = wave / 24, jt = wave % 24;  // all 4 waves of a block share mt
  unsigned phase = 0;

  {  // zero scratch region (hcat buffers, cell states)
    float* z = (float*)(ws + OFF_ZBEG);
    const int nz = (OFF_ZEND - OFF_ZBEG) / 4;
    for (int i = blockIdx.x * NTHR + tid; i < nz; i += NBLK * NTHR) z[i] = 0.0f;
  }
  gbar(bar, phase);
  if (blockIdx.x == 0 && tid < BATCH) {
    int b = tid;
    _Float16* dst = hAR + (size_t)b * KPA;
    const float* ip = inputs + ((size_t)b * 96 + 1) * 7;
#pragma unroll
    for (int j = 0; j < 7; ++j) dst[j] = (_Float16)ip[j];
    float p0 = inputs[((size_t)b * 96) * 7 + 0];
    dst[7] = (_Float16)p0;
    rAR[b * 96 + 0] = p0;
  }
  gbar(bar, phase);

  for (int t = 0; t < 95; ++t) {
    stage_A(sA, hAR, KPA, mt);
    Acc4 acc = wave_gemm(sA, KPA, war, KPA, 1, b0s, jt);
    cell0_tail(acc, mt, jt, hL1, HID, nullptr);
    gbar(bar, phase);

    stage_A(sA, hL1, HID, mt);
    acc = wave_gemm(sA, HID, wc1, KP1, 12, b1s, jt);
    cell0_tail(acc, mt, jt, nullptr, 0, h1f);
    gbar(bar, phase);

    if (blockIdx.x == 0 && tid < BATCH) {
      int b = tid;
      float s = b_out[0];
      const float* hr = h1f + b * HID;
      for (int k = 0; k < HID; ++k) s += hr[k] * W_out[k];
      rAR[b * 96 + t + 1] = s;
      if (t < 94) {
        _Float16* dst = hAR + (size_t)b * KPA;
        const float* ip = inputs + ((size_t)b * 96 + (t + 2)) * 7;
#pragma unroll
        for (int j = 0; j < 7; ++j) dst[j] = (_Float16)ip[j];
        dst[7] = (_Float16)s;
      }
    }
    gbar(bar, phase);
  }
  if (blockIdx.x == 0 && tid < BATCH) {
    float v = rAR[tid * 96 + 95];
    rFUT[tid * 16 + 0] = v;
    d_out[tid * HOR + 0] = v;
  }
}

// build mi[:, t, 0:8] for iteration n into hcat0 row (f16)
__device__ __forceinline__ void write_x(int n, int t, _Float16* hbase, int b,
                                        const float* __restrict__ inputs,
                                        const float* __restrict__ labels,
                                        const float* __restrict__ rAR,
                                        const float* __restrict__ rFUT) {
  float f[8];
  f[7] = rAR[b * 96 + t];
  if (t < 95 - n) {
    const float* ip = inputs + ((size_t)b * 96 + (t + n + 1)) * 7;
#pragma unroll
    for (int j = 0; j < 7; ++j) f[j] = ip[j];
  } else {
    int r = t - (95 - n);
    f[0] = rFUT[b * 16 + r];
    const float* lp = labels + ((size_t)b * HOR + r) * 7;
#pragma unroll
    for (int j = 1; j < 7; ++j) f[j] = lp[j];
  }
  _Float16* dst = hbase + (size_t)b * KP0;
#pragma unroll
  for (int j = 0; j < 8; ++j) dst[j] = (_Float16)f[j];
}

// ---------- kernel 3: one full 2-layer LSTM pass (iteration n) ----------
__global__ void __launch_bounds__(NTHR) lstm_iter_k(const float* __restrict__ inputs,
                                                    const float* __restrict__ labels,
                                                    const float* __restrict__ W_out,
                                                    const float* __restrict__ b_out,
                                                    float* __restrict__ d_out,
                                                    char* __restrict__ ws, int n, int barSlot) {
  __shared__ __align__(16) _Float16 sA[16 * KP1];

  unsigned* bar = (unsigned*)(ws + OFF_BAR) + barSlot;
  const _Float16* wc0 = (const _Float16*)(ws + OFF_WCAT0);
  const _Float16* wc1 = (const _Float16*)(ws + OFF_WCAT1);
  const float* b0s = (const float*)(ws + OFF_B0S);
  const float* b1s = (const float*)(ws + OFF_B1S);
  _Float16* hc0 = (_Float16*)(ws + OFF_HC0);
  _Float16* hc1 = (_Float16*)(ws + OFF_HC1);
  float* c0 = (float*)(ws + OFF_C0);
  float* c1 = (float*)(ws + OFF_C1);
  float* h1f = (float*)(ws + OFF_H1F);
  float* rAR = (float*)(ws + OFF_RAR);
  float* rFUT = (float*)(ws + OFF_RFUT);

  const int tid = threadIdx.x;
  const int wave = blockIdx.x * 4 + (tid >> 5);
  const int mt = wave / 24, jt = wave % 24;  // all 4 waves of a block share mt
  unsigned phase = 0;

  {  // zero states + hcat buffers (keeps K-padding columns at 0)
    float* z = (float*)(ws + OFF_ZBEG);
    const int nz = (OFF_ZEND - OFF_ZBEG) / 4;
    for (int i = blockIdx.x * NTHR + tid; i < nz; i += NBLK * NTHR) z[i] = 0.0f;
  }
  gbar(bar, phase);
  if (blockIdx.x == 0 && tid < BATCH) write_x(n, 0, hc0, tid, inputs, labels, rAR, rFUT);
  gbar(bar, phase);

  for (int t = 0; t < 96; ++t) {
    _Float16* A0  = hc0 + (size_t)(t & 1) * (BATCH * KP0);
    _Float16* A0n = hc0 + (size_t)((t + 1) & 1) * (BATCH * KP0);
    _Float16* A1  = hc1 + (size_t)(t & 1) * (BATCH * KP1);
    _Float16* A1n = hc1 + (size_t)((t + 1) & 1) * (BATCH * KP1);

    // layer 0: gates = [x_t | h0] @ [Wih0|Whh0]^T + b0
    stage_A(sA, A0, KP0, mt);
    Acc4 acc = wave_gemm(sA, KP0, wc0, KP0, 13, b0s, jt);
    lstm_tail(acc, mt, jt, c0, /*h0 -> GEMM1 A*/ A1, KP1, 0,
              /*h0 -> next GEMM0 A*/ A0n, KP0, 8, nullptr, 0);
    if (blockIdx.x == 0 && tid < BATCH && t < 95)
      write_x(n, t + 1, A0n, tid, inputs, labels, rAR, rFUT);
    gbar(bar, phase);

    // layer 1: gates = [h0_new | h1] @ [Wih1|Whh1]^T + b1
    stage_A(sA, A1, KP1, mt);
    acc = wave_gemm(sA, KP1, wc1, KP1, 24, b1s, jt);
    lstm_tail(acc, mt, jt, c1, /*h1 -> next GEMM1 A*/ A1n, KP1, HID,
              nullptr, 0, 0, h1f, (t == 95) ? 1 : 0);
    gbar(bar, phase);
  }

  if (blockIdx.x == 0 && tid < BATCH) {
    float s = b_out[0];
    const float* hr = h1f + tid * HID;
    for (int k = 0; k < HID; ++k) s += hr[k] * W_out[k];
    rFUT[tid * 16 + (n + 1)] = s;
    d_out[tid * HOR + (n + 1)] = s;
  }
}

extern "C" void kernel_launch(void* const* d_in, const int* in_sizes, int n_in,
                              void* d_out, int out_size, void* d_ws, size_t ws_size,
                              hipStream_t stream) {
  (void)in_sizes; (void)n_in; (void)out_size; (void)ws_size;
  const float* inputs = (const float*)d_in[0];
  const float* labels = (const float*)d_in[1];
  const float* Wih0 = (const float*)d_in[2];
  const float* Whh0 = (const float*)d_in[3];
  const float* bih0 = (const float*)d_in[4];
  const float* bhh0 = (const float*)d_in[5];
  const float* Wih1 = (const float*)d_in[6];
  const float* Whh1 = (const float*)d_in[7];
  const float* bih1 = (const float*)d_in[8];
  const float* bhh1 = (const float*)d_in[9];
  const float* W_out = (const float*)d_in[10];
  const float* b_out = (const float*)d_in[11];
  float* out = (float*)d_out;
  char* ws = (char*)d_ws;

  lstm_prep_k<<<1024, 256, 0, stream>>>(Wih0, Whh0, bih0, bhh0,
                                        Wih1, Whh1, bih1, bhh1, ws);
  lstm_ar_k<<<NBLK, NTHR, 0, stream>>>(inputs, W_out, b_out, out, ws, 0);
  for (int n = 0; n < 11; ++n)
    lstm_iter_k<<<NBLK, NTHR, 0, stream>>>(inputs, labels, W_out, b_out, out, ws, n, 1 + n);
}
